// ViT_13469017441054
// MI455X (gfx1250) — matrix operand order repaired and checked by reading, not compiled, not run on hardware
//
#include <hip/hip_runtime.h>

// ---------------------------------------------------------------------------
// ViT-Base forward for MI455X (gfx1250): all GEMMs via v_wmma_f32_16x16x32_bf16
// Round 2: vectorized (16B) LDS staging, 128x128 block tile (8 WMMA / K-step).
// ---------------------------------------------------------------------------

typedef __attribute__((ext_vector_type(16))) __bf16 v16bf;
typedef __attribute__((ext_vector_type(8)))  __bf16 v8bf;
typedef __attribute__((ext_vector_type(8)))  float  v8f;

namespace {
constexpr int NB     = 16;    // batch
constexpr int SEQ    = 197;
constexpr int DIM    = 768;
constexpr int HEADS  = 12;
constexpr int DHEAD  = 64;
constexpr int FFD    = 3072;
constexpr int NLAYER = 12;
constexpr int SPAD   = 224;   // SEQ padded to multiple of 32 for WMMA tiling
constexpr int NPATCH = 196;
constexpr int ROWS   = NB * SEQ;     // 3152 (multiple of 16)
constexpr int PROWS  = NB * NPATCH;  // 3136
constexpr float ATT_SCALE = 0.03608439182435161f;  // 1/sqrt(768)

constexpr int BM = 128, BN = 128, BK = 32;
constexpr int NFRAG = BN / 16;       // 8 WMMA issues per wave per K-step
}

static __device__ __forceinline__ v8bf zero8() {
  v8bf z;
  #pragma unroll
  for (int t = 0; t < 8; ++t) z[t] = (__bf16)0.f;
  return z;
}

// ---------------------------------------------------------------------------
// Tiled bf16 GEMM with WMMA.  C = A(MxK) * B(KxN) [+bias][+resid][GELU].
// A row-major [lda], B row-major [K,N] (or [N,K] if b_transposed).
// Batched via blockIdx.z; C offset = (z/hdiv)*sCb + (z%hdiv)*sCh.
// Rows >= m_valid are not written (lets us scatter padded attention tiles).
// Requires: K % 32 == 0, N % 8 == 0, lda/ldb element strides % 8 == 0.
// ---------------------------------------------------------------------------
__global__ __launch_bounds__(256) void k_gemm(
    const __bf16* __restrict__ A, const __bf16* __restrict__ Bm,
    float* __restrict__ Cf, __bf16* __restrict__ Cb,
    const float* __restrict__ bias, const float* __restrict__ resid,
    int M, int N, int K, int lda, int ldb, int ldc,
    long long sA, long long sB, long long sCb, long long sCh,
    int hdiv, int m_valid, int b_transposed, int gelu)
{
  // row stride 40 halves = 80B = 5*16B: 16B-aligned rows, conflict-free frags
  __shared__ __bf16 As[BM][BK + 8];
  __shared__ __bf16 Bs[BN][BK + 8];   // stored transposed: [n][k]

  const int z  = blockIdx.z;
  const int zb = z / hdiv, zh = z % hdiv;
  const __bf16* Ap = A  + (long long)z * sA;
  const __bf16* Bp = Bm + (long long)z * sB;
  const long long coff = (long long)zb * sCb + (long long)zh * sCh;

  const int m0 = blockIdx.y * BM;
  const int n0 = blockIdx.x * BN;

  const int tid  = threadIdx.x;
  const int wave = tid >> 5;      // 0..7, owns rows m0+16*wave..+15
  const int lane = tid & 31;
  const int lm   = lane & 15;
  const int lhi  = lane >> 4;

  v8f acc[NFRAG] = {};

  for (int k0 = 0; k0 < K; k0 += BK) {
    // ---- stage A tile (BM x BK) in 16B chunks: 512 chunks, 2 per thread
    #pragma unroll
    for (int i = tid; i < BM * BK / 8; i += 256) {
      const int m  = i >> 2;          // 4 chunks per row
      const int kc = (i & 3) << 3;    // 0,8,16,24
      const int gm = m0 + m;
      v8bf v = zero8();
      if (gm < M)
        v = *reinterpret_cast<const v8bf*>(Ap + (long long)gm * lda + (k0 + kc));
      *reinterpret_cast<v8bf*>(&As[m][kc]) = v;
    }
    // ---- stage B tile as [n][k]
    if (b_transposed) {
      // B stored [N,K]: contiguous along k -> straight 16B copy
      #pragma unroll
      for (int i = tid; i < BN * BK / 8; i += 256) {
        const int n  = i >> 2;
        const int kc = (i & 3) << 3;
        const int gn = n0 + n;
        v8bf v = zero8();
        if (gn < N)
          v = *reinterpret_cast<const v8bf*>(Bp + (long long)gn * ldb + (k0 + kc));
        *reinterpret_cast<v8bf*>(&Bs[n][kc]) = v;
      }
    } else {
      // B stored [K,N]: 16B load along n, 8-way scatter into [n][k]
      #pragma unroll
      for (int i = tid; i < BN * BK / 8; i += 256) {
        const int kk = i >> 4;            // BN/8 = 16 chunks per k-row
        const int nc = (i & 15) << 3;
        const int gn0 = n0 + nc;
        v8bf v = zero8();
        if (gn0 < N)
          v = *reinterpret_cast<const v8bf*>(Bp + (long long)(k0 + kk) * ldb + gn0);
        #pragma unroll
        for (int j = 0; j < 8; ++j) Bs[nc + j][kk] = v[j];
      }
    }
    // hint-prefetch next A tile (gfx1250 global_prefetch_b8)
    if (k0 + BK < K) {
      const int m  = tid >> 2;
      const int kc = (tid & 3) << 3;
      const int gm = m0 + m;
      if (gm < M) __builtin_prefetch(Ap + (long long)gm * lda + (k0 + BK + kc), 0, 0);
    }
    __syncthreads();

    // A fragment: lane lm holds row 16*wave+lm; lane-half lhi holds k = 16*lhi + t (contiguous; B uses same order)
    v16bf afrag;
    {
      const int row = (wave << 4) + lm;
      #pragma unroll
      for (int t = 0; t < 16; ++t) {
        int kk = (lhi << 4) + t;
        afrag[t] = As[row][kk];
      }
    }
    #pragma unroll
    for (int nf = 0; nf < NFRAG; ++nf) {
      v16bf bfrag;
      const int col   = (nf << 4) + lm;
      const int kbase = lhi << 4;
      #pragma unroll
      for (int t = 0; t < 16; ++t) bfrag[t] = Bs[col][kbase + t];
      acc[nf] = __builtin_amdgcn_wmma_f32_16x16x32_bf16(
          false, afrag, false, bfrag, (short)0, acc[nf], false, false);
    }
    __syncthreads();
  }

  // epilogue: acc[nf][r] -> C[m0 + 16*wave + 8*lhi + r][n0 + 16*nf + lm]
  #pragma unroll
  for (int nf = 0; nf < NFRAG; ++nf) {
    const int gn = n0 + (nf << 4) + lm;
    if (gn >= N) continue;
    const float bv = bias ? bias[gn] : 0.f;
    #pragma unroll
    for (int r = 0; r < 8; ++r) {
      const int gm = m0 + (wave << 4) + (lhi << 3) + r;
      if (gm >= M || gm >= m_valid) continue;
      float v = acc[nf][r] + bv;
      const long long ci = coff + (long long)gm * ldc + gn;
      if (resid) v += resid[ci];
      if (gelu)  v = 0.5f * v * (1.f + erff(v * 0.70710678118654752f));
      if (Cf) Cf[ci] = v;
      if (Cb) Cb[ci] = (__bf16)v;
    }
  }
}

// ---------------------------------------------------------------------------
// Patch embedding im2col: x[16,3,224,224] -> A0 bf16 [3136, 768]
// ---------------------------------------------------------------------------
__global__ __launch_bounds__(256) void k_im2col(const float* __restrict__ x,
                                                __bf16* __restrict__ A0)
{
  long long idx = (long long)blockIdx.x * 256 + threadIdx.x;
  if (idx >= (long long)PROWS * DIM) return;
  int col = idx % DIM;
  long long row = idx / DIM;
  int b = row / NPATCH, p = row % NPATCH;
  int ph = p / 14, pw = p % 14;
  int c = col >> 8, r = (col >> 4) & 15, cc = col & 15;
  A0[idx] = (__bf16)x[(((long long)b * 3 + c) * 224 + (ph * 16 + r)) * 224 + (pw * 16 + cc)];
}

// fp32 [R,C] -> bf16 [C,R] (for conv_w -> [in,out])
__global__ __launch_bounds__(256) void k_transpose(const float* __restrict__ in,
                                                   __bf16* __restrict__ out, int R, int C)
{
  long long idx = (long long)blockIdx.x * 256 + threadIdx.x;
  if (idx >= (long long)R * C) return;
  int c = idx % C;
  long long r = idx / C;
  out[(long long)c * R + r] = (__bf16)in[idx];
}

__global__ __launch_bounds__(256) void k_cvt_bf16(const float* __restrict__ in,
                                                  __bf16* __restrict__ out, long long n)
{
  long long idx = (long long)blockIdx.x * 256 + threadIdx.x;
  if (idx < n) out[idx] = (__bf16)in[idx];
}

// h[b,s,d] = (s==0 ? cls[d] : Pe[b*196+s-1, d]) + pos[s,d]
__global__ __launch_bounds__(256) void k_assemble(const float* __restrict__ Pe,
                                                  const float* __restrict__ cls,
                                                  const float* __restrict__ pos,
                                                  float* __restrict__ h)
{
  long long idx = (long long)blockIdx.x * 256 + threadIdx.x;
  if (idx >= (long long)ROWS * DIM) return;
  int d = idx % DIM;
  long long row = idx / DIM;
  int b = row / SEQ, s = row % SEQ;
  float v = (s == 0) ? cls[d] : Pe[((long long)(b * NPATCH + s - 1)) * DIM + d];
  h[idx] = v + pos[(long long)s * DIM + d];
}

// qkv fp32 [3152,768] -> per-head bf16 [B,H,SPAD,DHEAD], zero-padded rows
__global__ __launch_bounds__(256) void k_scatter_head(const float* __restrict__ src,
                                                      __bf16* __restrict__ dst)
{
  long long idx = (long long)blockIdx.x * 256 + threadIdx.x;
  const long long total = (long long)NB * HEADS * SPAD * DHEAD;
  if (idx >= total) return;
  int d = idx & 63;
  int s = (idx >> 6) % SPAD;
  int h = (int)((idx / ((long long)DHEAD * SPAD)) % HEADS);
  int b = (int)(idx / ((long long)DHEAD * SPAD * HEADS));
  __bf16 v = (__bf16)0.f;
  if (s < SEQ) v = (__bf16)src[((long long)(b * SEQ + s)) * DIM + h * DHEAD + d];
  dst[idx] = v;
}

// softmax over row of padded scores [B*H, SPAD, SPAD]; writes bf16 probs.
__global__ __launch_bounds__(256) void k_softmax(const float* __restrict__ S,
                                                 __bf16* __restrict__ P)
{
  __shared__ float red[256];
  const int row = blockIdx.x;          // bh*SPAD + s
  const int s   = row % SPAD;
  const long long base = (long long)row * SPAD;
  const int t = threadIdx.x;
  if (s >= SEQ) {                      // padded query rows: zero, never used
    if (t < SPAD) P[base + t] = (__bf16)0.f;
    return;
  }
  float v = (t < SEQ) ? S[base + t] * ATT_SCALE : -3.0e38f;
  red[t] = v; __syncthreads();
  for (int o = 128; o > 0; o >>= 1) { if (t < o) red[t] = fmaxf(red[t], red[t + o]); __syncthreads(); }
  const float mx = red[0]; __syncthreads();
  float e = (t < SEQ) ? expf(v - mx) : 0.f;
  red[t] = e; __syncthreads();
  for (int o = 128; o > 0; o >>= 1) { if (t < o) red[t] += red[t + o]; __syncthreads(); }
  const float inv = 1.f / red[0];
  if (t < SPAD) P[base + t] = (__bf16)(e * inv);   // cols >= SEQ get exact 0
}

// LayerNorm over D=768; in fp32 with row stride; writes fp32 and/or bf16.
__global__ __launch_bounds__(256) void k_layernorm(const float* __restrict__ in,
                                                   long long rowstride,
                                                   const float* __restrict__ g,
                                                   const float* __restrict__ b,
                                                   float* __restrict__ outf,
                                                   __bf16* __restrict__ outb,
                                                   long long ostride)
{
  __shared__ float red[256];
  const long long row = blockIdx.x;
  const float* p = in + row * rowstride;
  const int t = threadIdx.x;
  float s = 0.f;
  for (int d = t; d < DIM; d += 256) s += p[d];
  red[t] = s; __syncthreads();
  for (int o = 128; o > 0; o >>= 1) { if (t < o) red[t] += red[t + o]; __syncthreads(); }
  const float mean = red[0] / (float)DIM;
  __syncthreads();
  float vs = 0.f;
  for (int d = t; d < DIM; d += 256) { float x = p[d] - mean; vs += x * x; }
  red[t] = vs; __syncthreads();
  for (int o = 128; o > 0; o >>= 1) { if (t < o) red[t] += red[t + o]; __syncthreads(); }
  const float rstd = rsqrtf(red[0] / (float)DIM + 1e-5f);
  for (int d = t; d < DIM; d += 256) {
    float v = (p[d] - mean) * rstd * g[d] + b[d];
    if (outf) outf[row * ostride + d] = v;
    if (outb) outb[row * ostride + d] = (__bf16)v;
  }
}

// ---------------------------------------------------------------------------
// Host orchestration
// ---------------------------------------------------------------------------
static inline int cdiv(long long a, long long b) { return (int)((a + b - 1) / b); }

static void gemm(hipStream_t st, const __bf16* A, const __bf16* B,
                 float* Cf, __bf16* Cb, const float* bias, const float* resid,
                 int M, int N, int K, int lda, int ldb, int ldc,
                 long long sA, long long sB, long long sCb, long long sCh,
                 int hdiv, int batch, int m_valid, int btr, int gl)
{
  dim3 g(cdiv(N, BN), cdiv(M, BM), batch);
  k_gemm<<<g, 256, 0, st>>>(A, B, Cf, Cb, bias, resid, M, N, K, lda, ldb, ldc,
                            sA, sB, sCb, sCh, hdiv, m_valid, btr, gl);
}

extern "C" void kernel_launch(void* const* d_in, const int* in_sizes, int n_in,
                              void* d_out, int out_size, void* d_ws, size_t ws_size,
                              hipStream_t stream)
{
  const float* x      = (const float*)d_in[0];
  const float* conv_w = (const float*)d_in[1];
  const float* conv_b = (const float*)d_in[2];
  const float* cls    = (const float*)d_in[3];
  const float* pos    = (const float*)d_in[4];
  const float* ln1_s  = (const float*)d_in[5];
  const float* ln1_b  = (const float*)d_in[6];
  const float* wq     = (const float*)d_in[7];
  const float* bq     = (const float*)d_in[8];
  const float* wk     = (const float*)d_in[9];
  const float* bk     = (const float*)d_in[10];
  const float* wv     = (const float*)d_in[11];
  const float* bv     = (const float*)d_in[12];
  const float* wo     = (const float*)d_in[13];
  const float* bo     = (const float*)d_in[14];
  const float* ln2_s  = (const float*)d_in[15];
  const float* ln2_b  = (const float*)d_in[16];
  const float* w1     = (const float*)d_in[17];
  const float* b1     = (const float*)d_in[18];
  const float* w2     = (const float*)d_in[19];
  const float* b2     = (const float*)d_in[20];
  const float* lnf_s  = (const float*)d_in[21];
  const float* lnf_b  = (const float*)d_in[22];

  // ---- workspace carve-up (aligned, aliased where lifetimes don't overlap)
  char* base = (char*)d_ws;
  size_t off = 0;
  auto alloc = [&](size_t bytes) -> char* {
    char* p = base + off;
    off = (off + bytes + 255) & ~(size_t)255;
    return p;
  };
  float*  hbuf    = (float*) alloc((size_t)ROWS * DIM * 4);
  __bf16* hn      = (__bf16*)alloc((size_t)ROWS * DIM * 2);
  __bf16* hn2     = (__bf16*)alloc((size_t)ROWS * FFD * 2);
  float*  qtmp    = (float*) alloc((size_t)ROWS * DIM * 4);          // also Pe
  __bf16* wbuf    = (__bf16*)alloc((size_t)DIM * FFD * 2);
  __bf16* Qb      = (__bf16*)alloc((size_t)NB * HEADS * SPAD * DHEAD * 2);
  __bf16* Kb      = (__bf16*)alloc((size_t)NB * HEADS * SPAD * DHEAD * 2);
  __bf16* Vb      = (__bf16*)alloc((size_t)NB * HEADS * SPAD * DHEAD * 2);
  float*  scoresf = (float*) alloc((size_t)NB * HEADS * SPAD * SPAD * 4);
  __bf16* Pb      = (__bf16*)alloc((size_t)NB * HEADS * SPAD * SPAD * 2);
  __bf16* attnout = (__bf16*)alloc((size_t)ROWS * DIM * 2);
  // aliases (used only before attention buffers are live)
  __bf16* A0 = (__bf16*)scoresf;   // im2col matrix [3136,768]
  __bf16* Wc = (__bf16*)Pb;        // transposed conv weight [768,768]
  float*  Pe = qtmp;               // patch-embed GEMM output [3136,768]
  (void)ws_size; (void)in_sizes; (void)n_in; (void)out_size;

  const long long hstride = (long long)SPAD * DHEAD;   // per-(b,h) QKV stride
  const long long sstride = (long long)SPAD * SPAD;    // per-(b,h) score stride
  const int BH = NB * HEADS;                           // 192

  // ---- patch embedding as GEMM
  k_im2col   <<<cdiv((long long)PROWS * DIM, 256), 256, 0, stream>>>(x, A0);
  k_transpose<<<cdiv((long long)DIM * DIM, 256),   256, 0, stream>>>(conv_w, Wc, DIM, DIM);
  gemm(stream, A0, Wc, Pe, nullptr, conv_b, nullptr,
       PROWS, DIM, DIM, DIM, DIM, DIM, 0, 0, 0, 0, 1, 1, PROWS, 0, 0);
  k_assemble <<<cdiv((long long)ROWS * DIM, 256),  256, 0, stream>>>(Pe, cls, pos, hbuf);

  // ---- transformer blocks
  for (int l = 0; l < NLAYER; ++l) {
    const float* l1s = ln1_s + (long long)l * DIM;
    const float* l1b = ln1_b + (long long)l * DIM;
    const float* l2s = ln2_s + (long long)l * DIM;
    const float* l2b = ln2_b + (long long)l * DIM;
    const float* Wq = wq + (long long)l * DIM * DIM; const float* Bq = bq + (long long)l * DIM;
    const float* Wk = wk + (long long)l * DIM * DIM; const float* Bk = bk + (long long)l * DIM;
    const float* Wv = wv + (long long)l * DIM * DIM; const float* Bv = bv + (long long)l * DIM;
    const float* Wo = wo + (long long)l * DIM * DIM; const float* Bo = bo + (long long)l * DIM;
    const float* W1 = w1 + (long long)l * DIM * FFD; const float* B1 = b1 + (long long)l * FFD;
    const float* W2 = w2 + (long long)l * FFD * DIM; const float* B2 = b2 + (long long)l * DIM;

    // LN1 -> hn (bf16)
    k_layernorm<<<ROWS, 256, 0, stream>>>(hbuf, DIM, l1s, l1b, nullptr, hn, DIM);

    // Q, K, V projections + per-head scatter
    const float* Wqkv[3] = { Wq, Wk, Wv };
    const float* Bqkv[3] = { Bq, Bk, Bv };
    __bf16*      Dqkv[3] = { Qb, Kb, Vb };
    for (int j = 0; j < 3; ++j) {
      k_cvt_bf16<<<cdiv((long long)DIM * DIM, 256), 256, 0, stream>>>(Wqkv[j], wbuf, (long long)DIM * DIM);
      gemm(stream, hn, wbuf, qtmp, nullptr, Bqkv[j], nullptr,
           ROWS, DIM, DIM, DIM, DIM, DIM, 0, 0, 0, 0, 1, 1, ROWS, 0, 0);
      k_scatter_head<<<cdiv((long long)BH * SPAD * DHEAD, 256), 256, 0, stream>>>(qtmp, Dqkv[j]);
    }

    // scores = Q * K^T  (batched over B*H; B operand transposed)
    gemm(stream, Qb, Kb, scoresf, nullptr, nullptr, nullptr,
         SPAD, SPAD, DHEAD, DHEAD, DHEAD, SPAD,
         hstride, hstride, sstride, 0, 1, BH, SPAD, 1, 0);

    // softmax (applies 1/sqrt(768), zeroes padded cols/rows) -> Pb bf16
    k_softmax<<<BH * SPAD, 256, 0, stream>>>(scoresf, Pb);

    // O = P * V, written directly into attnout[b*197+s, h*64+d] as bf16
    gemm(stream, Pb, Vb, nullptr, attnout, nullptr, nullptr,
         SPAD, DHEAD, SPAD, SPAD, DHEAD, DIM,
         sstride, hstride, (long long)SEQ * DIM, DHEAD, HEADS, BH, SEQ, 0, 0);

    // out projection + residual -> hbuf
    k_cvt_bf16<<<cdiv((long long)DIM * DIM, 256), 256, 0, stream>>>(Wo, wbuf, (long long)DIM * DIM);
    gemm(stream, attnout, wbuf, hbuf, nullptr, Bo, hbuf,
         ROWS, DIM, DIM, DIM, DIM, DIM, 0, 0, 0, 0, 1, 1, ROWS, 0, 0);

    // LN2 -> hn
    k_layernorm<<<ROWS, 256, 0, stream>>>(hbuf, DIM, l2s, l2b, nullptr, hn, DIM);

    // FC1 + exact GELU -> hn2 (bf16)
    k_cvt_bf16<<<cdiv((long long)DIM * FFD, 256), 256, 0, stream>>>(W1, wbuf, (long long)DIM * FFD);
    gemm(stream, hn, wbuf, nullptr, hn2, B1, nullptr,
         ROWS, FFD, DIM, DIM, FFD, FFD, 0, 0, 0, 0, 1, 1, ROWS, 0, 1);

    // FC2 + residual -> hbuf
    k_cvt_bf16<<<cdiv((long long)FFD * DIM, 256), 256, 0, stream>>>(W2, wbuf, (long long)FFD * DIM);
    gemm(stream, hn2, wbuf, hbuf, nullptr, B2, hbuf,
         ROWS, DIM, FFD, FFD, DIM, DIM, 0, 0, 0, 0, 1, 1, ROWS, 0, 0);
  }

  // ---- final LN on CLS token rows only -> d_out [16,768] fp32
  k_layernorm<<<NB, 256, 0, stream>>>(hbuf, (long long)SEQ * DIM, lnf_s, lnf_b,
                                      (float*)d_out, nullptr, DIM);
}
